// SlotRouterGroup_90744069030176
// MI455X (gfx1250) — matrix operand order repaired and checked by reading, not compiled, not run on hardware
//
#include <hip/hip_runtime.h>

// ---------------------------------------------------------------------------
// Types for gfx1250 WMMA
// ---------------------------------------------------------------------------
typedef __bf16 bf16;
typedef __attribute__((ext_vector_type(16))) __bf16 bf16x16;
typedef __attribute__((ext_vector_type(8)))  __bf16 bf16x8;
typedef __attribute__((ext_vector_type(8)))  float  v8f;

#define DEV __device__ __forceinline__

// Problem constants (match reference)
#define Bn   32
#define Cn   256
#define Dd   1024        // d = T*L
#define Kslots 8
#define Hh   128
#define BC   (Bn*Cn)     // 8192
#define BK   (Bn*Kslots) // 256

// ---------------------------------------------------------------------------
// WMMA fragment loaders (CDNA5 16x16x32 bf16 layouts, ISA 7.12.2)
// A (16xK=32): lanes 0-15 hold M=lane, K {0..7,16..23}; lanes 16-31 K {8..15,24..31}
// B (K=32x16): lane holds column n=lane&15; lanes 0-15 K {0..7,16..23}, lanes 16-31 K {8..15,24..31}
// Since B[k,n] = W[n,k] (row-major W, K-contiguous rows), both loads are simple.
// ---------------------------------------------------------------------------
DEV bf16x16 load_fragA(const bf16* __restrict__ tile, int ld, int lane) {
  int r  = lane & 15;
  int kb = (lane & 16) ? 8 : 0;
  const bf16* p = tile + (size_t)r * ld + kb;
  bf16x8 c0 = *reinterpret_cast<const bf16x8*>(p);
  bf16x8 c1 = *reinterpret_cast<const bf16x8*>(p + 16);
  bf16x16 f;
#pragma unroll
  for (int i = 0; i < 8; ++i) { f[i] = c0[i]; f[i + 8] = c1[i]; }
  return f;
}

DEV bf16x16 load_fragB(const bf16* __restrict__ tile, int ld, int lane) {
  int n  = lane & 15;
  int kb = (lane & 16) ? 8 : 0;
  const bf16* p = tile + (size_t)n * ld + kb;
  bf16x8 c0 = *reinterpret_cast<const bf16x8*>(p);
  bf16x8 c1 = *reinterpret_cast<const bf16x8*>(p + 16);
  bf16x16 f;
#pragma unroll
  for (int i = 0; i < 8; ++i) { f[i] = c0[i]; f[i + 8] = c1[i]; }
  return f;
}

// ---------------------------------------------------------------------------
// Generic batched GEMM: out[m,n] = sum_k A[m,k]*W[n,k] + bias[n]
// A: (M,K) bf16 row-major, W: (N,K) bf16 row-major, out: (M,N) f32.
// One wave per block computes a 16x64 tile (A fragment reused 4x).
// ---------------------------------------------------------------------------
__global__ void k_gemm_bf16(const bf16* __restrict__ A, const bf16* __restrict__ W,
                            const float* __restrict__ bias, float* __restrict__ out,
                            int M, int N, int K,
                            long long sA, long long sW, long long sO) {
  int lane = (int)threadIdx.x;     // blockDim.x == 32 (wave32)
  int m0 = blockIdx.x * 16;
  int n0 = blockIdx.y * 64;
  int b  = blockIdx.z;
  const bf16* Ab = A + (long long)b * sA;
  const bf16* Wb = W + (long long)b * sW;
  float*      Ob = out + (long long)b * sO;

  v8f acc0 = {}, acc1 = {}, acc2 = {}, acc3 = {};
  for (int k = 0; k < K; k += 32) {
    bf16x16 a = load_fragA(Ab + (size_t)m0 * K + k, K, lane);
    bf16x16 w0 = load_fragB(Wb + (size_t)(n0 +  0) * K + k, K, lane);
    bf16x16 w1 = load_fragB(Wb + (size_t)(n0 + 16) * K + k, K, lane);
    bf16x16 w2 = load_fragB(Wb + (size_t)(n0 + 32) * K + k, K, lane);
    bf16x16 w3 = load_fragB(Wb + (size_t)(n0 + 48) * K + k, K, lane);
    acc0 = __builtin_amdgcn_wmma_f32_16x16x32_bf16(false, a, false, w0, (short)0, acc0, false, false);
    acc1 = __builtin_amdgcn_wmma_f32_16x16x32_bf16(false, a, false, w1, (short)0, acc1, false, false);
    acc2 = __builtin_amdgcn_wmma_f32_16x16x32_bf16(false, a, false, w2, (short)0, acc2, false, false);
    acc3 = __builtin_amdgcn_wmma_f32_16x16x32_bf16(false, a, false, w3, (short)0, acc3, false, false);
  }
  int col   = lane & 15;
  int rbase = (lane & 16) ? 8 : 0;
  v8f accs[4] = {acc0, acc1, acc2, acc3};
#pragma unroll
  for (int t = 0; t < 4; ++t) {
    int n = n0 + 16 * t + col;
    float bv = bias ? bias[n] : 0.f;
#pragma unroll
    for (int r = 0; r < 8; ++r)
      Ob[(size_t)(m0 + rbase + r) * N + n] = accs[t][r] + bv;
  }
}

// ---------------------------------------------------------------------------
// Elementwise / reduction helpers (f32 glue)
// ---------------------------------------------------------------------------
__global__ void k_f32_to_bf16(const float* __restrict__ in, bf16* __restrict__ out, int n) {
  int i = blockIdx.x * 256 + threadIdx.x;
  if (i < n) out[i] = (bf16)in[i];
}

__global__ void k_relu_to_bf16(const float* __restrict__ in, bf16* __restrict__ out, int n) {
  int i = blockIdx.x * 256 + threadIdx.x;
  if (i < n) out[i] = (bf16)fmaxf(in[i], 0.f);
}

__global__ void k_add_inplace(float* __restrict__ dst, const float* __restrict__ src, int n) {
  int i = blockIdx.x * 256 + threadIdx.x;
  if (i < n) dst[i] += src[i];
}

// LayerNorm over rows of width 1024, output bf16.  One 256-thread block per row.
__global__ void k_ln_bf16(const float* __restrict__ x, const float* __restrict__ g,
                          const float* __restrict__ b, bf16* __restrict__ out) {
  __shared__ float red[256];
  int row = blockIdx.x, t = threadIdx.x;
  const float* xr = x + (size_t)row * Dd;
  float v4[4]; float s = 0.f;
#pragma unroll
  for (int c = 0; c < 4; ++c) { v4[c] = xr[t + 256 * c]; s += v4[c]; }
  red[t] = s; __syncthreads();
  for (int o = 128; o > 0; o >>= 1) { if (t < o) red[t] += red[t + o]; __syncthreads(); }
  float mean = red[0] * (1.f / Dd); __syncthreads();
  float s2 = 0.f;
#pragma unroll
  for (int c = 0; c < 4; ++c) { float d = v4[c] - mean; s2 += d * d; }
  red[t] = s2; __syncthreads();
  for (int o = 128; o > 0; o >>= 1) { if (t < o) red[t] += red[t + o]; __syncthreads(); }
  float rstd = rsqrtf(red[0] * (1.f / Dd) + 1e-5f);
#pragma unroll
  for (int c = 0; c < 4; ++c) {
    int col = t + 256 * c;
    out[(size_t)row * Dd + col] = (bf16)((v4[c] - mean) * rstd * g[col] + b[col]);
  }
}

// slots = mu + exp(logsigma) * noise
__global__ void k_slots_init(const float* __restrict__ mu, const float* __restrict__ ls,
                             const float* __restrict__ noise, float* __restrict__ slots) {
  int i = blockIdx.x * 256 + threadIdx.x;       // 0 .. BK*Dd-1
  int dd = i & (Dd - 1);
  slots[i] = mu[dd] + expf(ls[dd]) * noise[i];
}

// dots[b,i,j] = scale * q[b,i,:]·kk[b,j,:]; softmax over i (K=8); +1e-8.
// One block per (b,j).
__global__ void k_dots_softmax(const float* __restrict__ q, const float* __restrict__ kk,
                               float* __restrict__ attn) {
  __shared__ float red[256];
  __shared__ float s8[Kslots];
  int bj = blockIdx.x; int b = bj >> 8; int j = bj & 255; int t = threadIdx.x;
  const float* kr = kk + ((size_t)b * Cn + j) * Dd;
  float k4[4];
#pragma unroll
  for (int c = 0; c < 4; ++c) k4[c] = kr[t + 256 * c];
  for (int i = 0; i < Kslots; ++i) {
    const float* qr = q + ((size_t)b * Kslots + i) * Dd;
    float s = 0.f;
#pragma unroll
    for (int c = 0; c < 4; ++c) s += qr[t + 256 * c] * k4[c];
    red[t] = s; __syncthreads();
    for (int o = 128; o > 0; o >>= 1) { if (t < o) red[t] += red[t + o]; __syncthreads(); }
    if (t == 0) s8[i] = red[0] * 0.03125f;   // scale = 1024^-0.5
    __syncthreads();
  }
  if (t == 0) {
    float mx = s8[0];
    for (int i = 1; i < Kslots; ++i) mx = fmaxf(mx, s8[i]);
    float e[Kslots]; float sum = 0.f;
    for (int i = 0; i < Kslots; ++i) { e[i] = expf(s8[i] - mx); sum += e[i]; }
    for (int i = 0; i < Kslots; ++i)
      attn[((size_t)b * Kslots + i) * Cn + j] = e[i] / sum + 1e-8f;
  }
}

// attn row-normalize over j (C=256).  One block per (b,i).
__global__ void k_attn_norm(float* __restrict__ attn, float eps) {
  __shared__ float red[256];
  int row = blockIdx.x, t = threadIdx.x;
  float v = attn[(size_t)row * Cn + t];
  red[t] = v; __syncthreads();
  for (int o = 128; o > 0; o >>= 1) { if (t < o) red[t] += red[t + o]; __syncthreads(); }
  attn[(size_t)row * Cn + t] = v / (red[0] + eps);
}

// upd[b,i,:] = sum_j attn[b,i,j] * vv[b,j,:]  -> bf16 (GRU GEMM input)
__global__ void k_upd(const float* __restrict__ attn, const float* __restrict__ vv,
                      bf16* __restrict__ upd_bf) {
  __shared__ float sa[256];
  int bi = blockIdx.x; int b = bi >> 3; int t = threadIdx.x;
  sa[t] = attn[(size_t)bi * Cn + t]; __syncthreads();
  float acc[4] = {0.f, 0.f, 0.f, 0.f};
  const float* vb = vv + (size_t)b * Cn * Dd;
  for (int j = 0; j < Cn; ++j) {
    float a = sa[j];
    const float* vr = vb + (size_t)j * Dd;
#pragma unroll
    for (int c = 0; c < 4; ++c) acc[c] += a * vr[t + 256 * c];
  }
#pragma unroll
  for (int c = 0; c < 4; ++c) upd_bf[(size_t)bi * Dd + t + 256 * c] = (bf16)acc[c];
}

// GRU combine; updates slots in place (h read before write per-thread).
__global__ void k_gru(const float* __restrict__ gi, const float* __restrict__ gh,
                      float* __restrict__ slots) {
  int i = blockIdx.x * 256 + threadIdx.x;       // BK*Dd
  int row = i >> 10; int col = i & 1023;
  size_t base = (size_t)row * (3 * Dd);
  float ir = gi[base + col],          hr = gh[base + col];
  float iz = gi[base + Dd + col],     hz = gh[base + Dd + col];
  float ic = gi[base + 2 * Dd + col], hc = gh[base + 2 * Dd + col];
  float r = 1.f / (1.f + expf(-(ir + hr)));
  float z = 1.f / (1.f + expf(-(iz + hz)));
  float cand = tanhf(ic + r * hc);
  float h = slots[i];
  slots[i] = (1.f - z) * cand + z * h;
}

// WW[g,d2,j,c] = sum_dd Wbr[g,d2,j*32+dd] * Wp[dd,c]
__global__ void k_ww(const float* __restrict__ Wbr, const float* __restrict__ Wp,
                     float* __restrict__ WW) {
  int i = blockIdx.x * 256 + threadIdx.x;       // 65536
  int c = i & 255; int j = (i >> 8) & 1; int d2 = (i >> 9) & 31; int g = i >> 14;
  float s = 0.f;
  const float* wb = Wbr + ((size_t)(g * 32 + d2)) * 64 + j * 32;
#pragma unroll
  for (int dd = 0; dd < 32; ++dd) s += wb[dd] * Wp[(size_t)dd * Cn + c];
  WW[i] = s;
}

// Meff[b,gd2,c] = sum_j relu(gate[2g+j]) * attn[b,2g+j,c] * WW[g,d2,j,c]  (bf16)
__global__ void k_meff(const float* __restrict__ attn, const float* __restrict__ WW,
                       const float* __restrict__ gates, bf16* __restrict__ Meff) {
  int i = blockIdx.x * 256 + threadIdx.x;       // 1,048,576 = (b,g,d2,c)
  int c = i & 255; int d2 = (i >> 8) & 31; int g = (i >> 13) & 3; int b = i >> 15;
  float s = 0.f;
#pragma unroll
  for (int j = 0; j < 2; ++j) {
    int k = g * 2 + j;
    float gt = fmaxf(gates[k], 0.f);
    s += gt * attn[((size_t)b * Kslots + k) * Cn + c]
            * WW[(((size_t)(g * 32 + d2)) * 2 + j) * Cn + c];
  }
  Meff[i] = (bf16)s;
}

// xT[b,tl,c] = bf16(x[b,c,tl])  (32x32 LDS-tiled transpose)
__global__ void k_xT(const float* __restrict__ x, bf16* __restrict__ xT) {
  __shared__ float tile[32][33];
  int b = blockIdx.z;
  int tl0 = blockIdx.x * 32, c0 = blockIdx.y * 32;
  int tx = threadIdx.x, ty = threadIdx.y;       // block (32,8)
#pragma unroll
  for (int r = 0; r < 4; ++r) {
    int c = c0 + ty + 8 * r;
    tile[ty + 8 * r][tx] = x[((size_t)b * Cn + c) * Dd + tl0 + tx];
  }
  __syncthreads();
#pragma unroll
  for (int r = 0; r < 4; ++r) {
    int tl = tl0 + ty + 8 * r;
    xT[((size_t)b * Dd + tl) * Cn + c0 + tx] = (bf16)tile[tx][ty + 8 * r];
  }
}

__global__ void k_gelu(float* __restrict__ y, int n) {
  int i = blockIdx.x * 256 + threadIdx.x;
  if (i < n) { float v = y[i]; y[i] = 0.5f * v * (1.f + erff(v * 0.70710678f)); }
}

// BatchNorm over (b,tl) per channel; y layout (B,128,1024).  One block per ch.
__global__ void k_bn(const float* __restrict__ y, const float* __restrict__ g,
                     const float* __restrict__ bta, float* __restrict__ out) {
  __shared__ float r1[256], r2[256];
  int ch = blockIdx.x, t = threadIdx.x;
  float s = 0.f, s2 = 0.f;
  for (int bb = 0; bb < Bn; ++bb) {
    const float* p = y + (size_t)bb * 131072 + (size_t)ch * 1024;
#pragma unroll
    for (int c = 0; c < 4; ++c) { float v = p[t + 256 * c]; s += v; s2 += v * v; }
  }
  r1[t] = s; r2[t] = s2; __syncthreads();
  for (int o = 128; o > 0; o >>= 1) {
    if (t < o) { r1[t] += r1[t + o]; r2[t] += r2[t + o]; }
    __syncthreads();
  }
  float mean = r1[0] * (1.f / 32768.f);
  float var  = r2[0] * (1.f / 32768.f) - mean * mean;
  float rstd = rsqrtf(var + 1e-5f);
  float gg = g[ch], bt = bta[ch];
  for (int bb = 0; bb < Bn; ++bb) {
    const float* p = y  + (size_t)bb * 131072 + (size_t)ch * 1024;
    float*      o2 = out + (size_t)bb * 131072 + (size_t)ch * 1024;
#pragma unroll
    for (int c = 0; c < 4; ++c) { int idx = t + 256 * c; o2[idx] = (p[idx] - mean) * rstd * gg + bt; }
  }
}

// ---------------------------------------------------------------------------
// Host launcher
// ---------------------------------------------------------------------------
extern "C" void kernel_launch(void* const* d_in, const int* in_sizes, int n_in,
                              void* d_out, int out_size, void* d_ws, size_t ws_size,
                              hipStream_t stream) {
  (void)in_sizes; (void)n_in; (void)out_size; (void)ws_size;
  const float* x    = (const float*)d_in[0];
  const float* nz   = (const float*)d_in[1];
  const float* mu   = (const float*)d_in[2];
  const float* lsig = (const float*)d_in[3];
  const float* Wq   = (const float*)d_in[4];  const float* bq  = (const float*)d_in[5];
  const float* Wk   = (const float*)d_in[6];  const float* bk  = (const float*)d_in[7];
  const float* Wv   = (const float*)d_in[8];  const float* bv  = (const float*)d_in[9];
  const float* Wih  = (const float*)d_in[10]; const float* bih = (const float*)d_in[11];
  const float* Whh  = (const float*)d_in[12]; const float* bhh = (const float*)d_in[13];
  const float* W1   = (const float*)d_in[14]; const float* b1  = (const float*)d_in[15];
  const float* W2   = (const float*)d_in[16]; const float* b2  = (const float*)d_in[17];
  const float* ling = (const float*)d_in[18]; const float* linb = (const float*)d_in[19];
  const float* lnsg = (const float*)d_in[20]; const float* lnsb = (const float*)d_in[21];
  const float* lnfg = (const float*)d_in[22]; const float* lnfb = (const float*)d_in[23];
  const float* Wp   = (const float*)d_in[24];
  const float* gates= (const float*)d_in[25];
  const float* Wbr  = (const float*)d_in[26];
  const float* bng  = (const float*)d_in[27]; const float* bnb = (const float*)d_in[28];
  float* out = (float*)d_out;

  // --- workspace carve-up (256B aligned) ---
  char* ws = (char*)d_ws;
  size_t off = 0;
  auto alloc = [&](size_t bytes) -> void* {
    off = (off + 255) & ~(size_t)255;
    void* p = ws + off; off += bytes; return p;
  };
  bf16* inp_bf  = (bf16*)alloc((size_t)BC * Dd * 2);
  bf16* Wq_bf   = (bf16*)alloc((size_t)Dd * Dd * 2);
  bf16* Wk_bf   = (bf16*)alloc((size_t)Dd * Dd * 2);
  bf16* Wv_bf   = (bf16*)alloc((size_t)Dd * Dd * 2);
  bf16* Wih_bf  = (bf16*)alloc((size_t)3 * Dd * Dd * 2);
  bf16* Whh_bf  = (bf16*)alloc((size_t)3 * Dd * Dd * 2);
  bf16* W1_bf   = (bf16*)alloc((size_t)Hh * Dd * 2);
  bf16* W2_bf   = (bf16*)alloc((size_t)Dd * Hh * 2);
  float* kk     = (float*)alloc((size_t)BC * Dd * 4);
  float* vv     = (float*)alloc((size_t)BC * Dd * 4);
  float* slots  = (float*)alloc((size_t)BK * Dd * 4);
  float* qbuf   = (float*)alloc((size_t)BK * Dd * 4);
  float* attn   = (float*)alloc((size_t)Bn * Kslots * Cn * 4);
  bf16* lnS_bf  = (bf16*)alloc((size_t)BK * Dd * 2);
  bf16* h_bf    = (bf16*)alloc((size_t)BK * Dd * 2);
  bf16* ff_bf   = (bf16*)alloc((size_t)BK * Dd * 2);
  bf16* upd_bf  = (bf16*)alloc((size_t)BK * Dd * 2);
  float* gi     = (float*)alloc((size_t)BK * 3 * Dd * 4);
  float* gh     = (float*)alloc((size_t)BK * 3 * Dd * 4);
  float* mlp1   = (float*)alloc((size_t)BK * Hh * 4);
  bf16* mlp1_bf = (bf16*)alloc((size_t)BK * Hh * 2);
  float* mlp2   = (float*)alloc((size_t)BK * Dd * 4);
  float* WW     = (float*)alloc((size_t)4 * 32 * 2 * Cn * 4);
  bf16* Meff_bf = (bf16*)alloc((size_t)Bn * 128 * Cn * 2);
  bf16* xT_bf   = (bf16*)alloc((size_t)Bn * Dd * Cn * 2);
  float* ybuf   = (float*)alloc((size_t)Bn * 128 * Dd * 4);

  auto conv = [&](const float* s, bf16* d, int n) {
    k_f32_to_bf16<<<(n + 255) / 256, 256, 0, stream>>>(s, d, n);
  };
  auto gemm = [&](const bf16* A, const bf16* W, const float* bias, float* O,
                  int M, int N, int K, int batch, long long sA, long long sW, long long sO) {
    dim3 grid(M / 16, N / 64, batch);
    k_gemm_bf16<<<grid, 32, 0, stream>>>(A, W, bias, O, M, N, K, sA, sW, sO);
  };

  // --- weight conversion + token pre-processing ---
  conv(Wq, Wq_bf, Dd * Dd);
  conv(Wk, Wk_bf, Dd * Dd);
  conv(Wv, Wv_bf, Dd * Dd);
  conv(Wih, Wih_bf, 3 * Dd * Dd);
  conv(Whh, Whh_bf, 3 * Dd * Dd);
  conv(W1, W1_bf, Hh * Dd);
  conv(W2, W2_bf, Dd * Hh);
  k_ln_bf16<<<BC, 256, 0, stream>>>(x, ling, linb, inp_bf);
  gemm(inp_bf, Wk_bf, bk, kk, BC, Dd, Dd, 1, 0, 0, 0);
  gemm(inp_bf, Wv_bf, bv, vv, BC, Dd, Dd, 1, 0, 0, 0);
  k_slots_init<<<BK * Dd / 256, 256, 0, stream>>>(mu, lsig, nz, slots);
  k_xT<<<dim3(32, 8, Bn), dim3(32, 8), 0, stream>>>(x, xT_bf);
  k_ww<<<256, 256, 0, stream>>>(Wbr, Wp, WW);

  // --- slot-attention iterations ---
  for (int it = 0; it < 3; ++it) {
    k_ln_bf16<<<BK, 256, 0, stream>>>(slots, lnsg, lnsb, lnS_bf);
    gemm(lnS_bf, Wq_bf, bq, qbuf, BK, Dd, Dd, 1, 0, 0, 0);
    k_dots_softmax<<<BC, 256, 0, stream>>>(qbuf, kk, attn);
    k_attn_norm<<<BK, 256, 0, stream>>>(attn, 0.f);
    k_upd<<<BK, 256, 0, stream>>>(attn, vv, upd_bf);
    k_f32_to_bf16<<<BK * Dd / 256, 256, 0, stream>>>(slots, h_bf, BK * Dd);
    gemm(upd_bf, Wih_bf, bih, gi, BK, 3 * Dd, Dd, 1, 0, 0, 0);
    gemm(h_bf,   Whh_bf, bhh, gh, BK, 3 * Dd, Dd, 1, 0, 0, 0);
    k_gru<<<BK * Dd / 256, 256, 0, stream>>>(gi, gh, slots);
    k_ln_bf16<<<BK, 256, 0, stream>>>(slots, lnfg, lnfb, ff_bf);
    gemm(ff_bf, W1_bf, b1, mlp1, BK, Hh, Dd, 1, 0, 0, 0);
    k_relu_to_bf16<<<BK * Hh / 256, 256, 0, stream>>>(mlp1, mlp1_bf, BK * Hh);
    gemm(mlp1_bf, W2_bf, b2, mlp2, BK, Dd, Hh, 1, 0, 0, 0);
    k_add_inplace<<<BK * Dd / 256, 256, 0, stream>>>(slots, mlp2, BK * Dd);
  }

  // --- fused router + branch conv: y[b] = Meff[b] @ x[b] ---
  k_attn_norm<<<BK, 256, 0, stream>>>(attn, 1e-9f);
  k_meff<<<Bn * 128 * Cn / 256, 256, 0, stream>>>(attn, WW, gates, Meff_bf);
  gemm(Meff_bf, xT_bf, nullptr, ybuf, 128, Dd, Cn, Bn,
       (long long)128 * Cn, (long long)Dd * Cn, (long long)128 * Dd);
  k_gelu<<<Bn * 128 * Dd / 256, 256, 0, stream>>>(ybuf, Bn * 128 * Dd);
  k_bn<<<128, 256, 0, stream>>>(ybuf, bng, bnb, out);
}
